// Model_76879914598800
// MI455X (gfx1250) — matrix operand order repaired and checked by reading, not compiled, not run on hardware
//
#include <hip/hip_runtime.h>
#include <cstdint>
#include <cstddef>

// ---------------------------------------------------------------------------
// Types
// ---------------------------------------------------------------------------
typedef _Float16 v16h __attribute__((ext_vector_type(16)));
typedef _Float16 v8h  __attribute__((ext_vector_type(8)));
typedef float    v8f  __attribute__((ext_vector_type(8)));

#define PI_F 3.14159265358979323846f

enum { M_EDGE = 0, M_NODE = 1, M_U = 2 };

__device__ __forceinline__ float sspf(float x) {
    // softplus(x) - log(2), numerically stable
    return fmaxf(x, 0.0f) + log1pf(expf(-fabsf(x))) - 0.69314718055994531f;
}
__device__ __forceinline__ float sigmf(float x) {
    return 1.0f / (1.0f + expf(-x));
}

// Cooperative global->LDS copy, 16B granules (halves must be multiple of 8,
// both pointers 16B aligned).
__device__ __forceinline__ void copy_w(const _Float16* __restrict__ src,
                                       _Float16* dst, int halves, int tid) {
    const uint4* s = (const uint4*)src;
    uint4* d = (uint4*)dst;
    for (int i = tid; i < halves / 8; i += 256) d[i] = s[i];
}

// ---------------------------------------------------------------------------
// WMMA helpers. 16x16x32 f16 -> f32. A from LDS (row-major tile), B from an
// LDS-staged packed buffer laid out per-fragment [lane][16 halves].
//
// A layout (matches B): lane l<16 -> row l, K 0..15 sequential;
// lane l>=16 -> row l-16, K 16..31. Two ds_load_b128 per lane.
// B layout (contiguous k order, same as A): lane j<16 -> col j, K 0..15 sequential;
// lane j>=16 -> col j-16, K 16..31.
// ---------------------------------------------------------------------------
template<int DINL, int DOUTL>
__device__ __forceinline__ v8f wmma_rowtile(const _Float16* Xin,
                                            const _Float16* Bp,
                                            int wave, int lane, int nt)
{
    v8f acc = {};
    const int row  = wave * 16 + (lane & 15);
    const int kofs = (lane & 16) ? 16 : 0;
    const _Float16* abase = Xin + row * DINL + kofs;
    const _Float16* bbase = Bp + nt * 512 + lane * 16;
#pragma unroll
    for (int kb = 0; kb < DINL / 32; ++kb) {
        const _Float16* pa = abase + kb * 32;
        v8h alo = *(const v8h*)pa;
        v8h ahi = *(const v8h*)(pa + 8);
        v16h a = __builtin_shufflevector(alo, ahi,
                 0,1,2,3,4,5,6,7,8,9,10,11,12,13,14,15);
        const _Float16* pb = bbase + kb * (DOUTL / 16) * 512;
        v8h blo = *(const v8h*)pb;
        v8h bhi = *(const v8h*)(pb + 8);
        v16h b = __builtin_shufflevector(blo, bhi,
                 0,1,2,3,4,5,6,7,8,9,10,11,12,13,14,15);
        acc = __builtin_amdgcn_wmma_f32_16x16x32_f16(
                  false, a, false, b, (short)0, acc, false, false);
    }
    return acc;
}

// One full layer: DINL -> DOUTL, softplus, result to LDS as f16.
template<int DINL, int DOUTL>
__device__ __forceinline__ void layer_lds(const _Float16* Xin, _Float16* Xout,
                                          const _Float16* Bp,
                                          const float* bias,
                                          int wave, int lane)
{
#pragma unroll
    for (int nt = 0; nt < DOUTL / 16; ++nt) {
        v8f acc = wmma_rowtile<DINL, DOUTL>(Xin, Bp, wave, lane, nt);
        float bv = bias[nt * 16 + (lane & 15)];
        int mrow = wave * 16 + ((lane & 16) ? 8 : 0);
        int col  = nt * 16 + (lane & 15);
#pragma unroll
        for (int r = 0; r < 8; ++r)
            Xout[(mrow + r) * DOUTL + col] = (_Float16)sspf(acc[r] + bv);
    }
}

// ---------------------------------------------------------------------------
// Pre-MLP: src[M,32] f32 -> (32->64->32) -> out[M,32] f32.
// Weights staged in LDS: W1 = 2048 halves, W2 = 2048 halves.
// ---------------------------------------------------------------------------
template<int DIN, int D1, int D2>
__global__ __launch_bounds__(256)
void mlp2_pre(const float* __restrict__ src,
              const _Float16* __restrict__ w1, const float* __restrict__ b1,
              const _Float16* __restrict__ w2, const float* __restrict__ b2,
              float* __restrict__ out)
{
    __shared__ alignas(16) _Float16 Xa[128 * DIN];
    __shared__ alignas(16) _Float16 Xb[128 * D1];
    __shared__ alignas(16) _Float16 Wl[DIN * (D1 / 16) * 16 + D1 * (D2 / 16) * 16];
    __shared__ float Bl[D1 + D2];
    constexpr int W1H = DIN * D1;   // halves
    constexpr int W2H = D1 * D2;
    const int tid = threadIdx.x, wave = tid >> 5, lane = tid & 31;
    const int row0 = blockIdx.x * 128;

    copy_w(w1, Wl, W1H, tid);
    copy_w(w2, Wl + W1H, W2H, tid);
    if (tid < D1)              Bl[tid] = b1[tid];
    else if (tid < D1 + D2)    Bl[tid] = b2[tid - D1];

    for (int i = tid; i < 128 * DIN; i += 256)
        Xa[i] = (_Float16)src[(size_t)row0 * DIN + i];
    __syncthreads();

    layer_lds<DIN, D1>(Xa, Xb, Wl, Bl, wave, lane);
    __syncthreads();
#pragma unroll
    for (int nt = 0; nt < D2 / 16; ++nt) {
        v8f acc = wmma_rowtile<D1, D2>(Xb, Wl + W1H, wave, lane, nt);
        float bv = Bl[D1 + nt * 16 + (lane & 15)];
        int col = nt * 16 + (lane & 15);
        int mbase = wave * 16 + ((lane & 16) ? 8 : 0);
#pragma unroll
        for (int r = 0; r < 8; ++r) {
            int row = row0 + mbase + r;
            out[(size_t)row * D2 + col] = sspf(acc[r] + bv);
        }
    }
}

// ---------------------------------------------------------------------------
// Fused 3-layer phi MLP: gather/concat -> (DIN->64->64->32) -> fused
// residual write + atomic segment sums. Weights LDS-resident.
// ---------------------------------------------------------------------------
struct Mlp3Args {
    const float* src0; const float* src1; const float* src2;
    const int*   eidx; int div;
    const _Float16* w1; const float* b1;
    const _Float16* w2; const float* b2;
    const _Float16* w3; const float* b3;
    float* out;          // EDGE: new master (= resid + y); NODE/U: master (+= y)
    const float* resid;  // EDGE only
    float* acc0;         // EDGE: ev_acc[i0]; NODE: vu_acc[g]
    float* acc1;         // EDGE: eu_acc[g]
};

template<int MODE, int DIN>
__global__ __launch_bounds__(256)
void mlp3_kernel(Mlp3Args a)
{
    __shared__ alignas(16) _Float16 Xa[128 * DIN];
    __shared__ alignas(16) _Float16 Xb[128 * 64];
    __shared__ alignas(16) _Float16 Wl[DIN * 64 + 4096 + 2048];
    __shared__ float Bl[160];
    constexpr int W1H = DIN * 64;
    const int tid = threadIdx.x, wave = tid >> 5, lane = tid & 31;
    const int row0 = blockIdx.x * 128;

    // stage all three layers' packed weights + biases into LDS once
    copy_w(a.w1, Wl, W1H, tid);
    copy_w(a.w2, Wl + W1H, 4096, tid);
    copy_w(a.w3, Wl + W1H + 4096, 2048, tid);
    if (tid < 64)       Bl[tid] = a.b1[tid];
    else if (tid < 128) Bl[tid] = a.b2[tid - 64];
    else if (tid < 160) Bl[tid] = a.b3[tid - 128];

    if (MODE == M_EDGE) {
        // concat [v[i0] | e | v[i1] | u[g]]  (4 x 32 = 128)
        for (int i = tid; i < 128 * 128; i += 256) {
            int m = i >> 7, k = i & 127;
            int row = row0 + m;
            int sel = k >> 5, c = k & 31;
            float v;
            if (sel == 0)      v = a.src0[(size_t)a.eidx[2 * row]     * 32 + c];
            else if (sel == 1) v = a.src1[(size_t)row * 32 + c];
            else if (sel == 2) v = a.src0[(size_t)a.eidx[2 * row + 1] * 32 + c];
            else               v = a.src2[(size_t)(row / a.div) * 32 + c];
            Xa[i] = (_Float16)v;
        }
    } else {
        // concat [src0[row] | src1[row] | src2[row/div]]  (3 x 32 = 96)
        for (int i = tid; i < 128 * 96; i += 256) {
            int m = i / 96, k = i - m * 96;
            int row = row0 + m;
            int sel = k >> 5, c = k & 31;
            float v = (sel == 0) ? a.src0[(size_t)row * 32 + c]
                    : (sel == 1) ? a.src1[(size_t)row * 32 + c]
                                 : a.src2[(size_t)(row / a.div) * 32 + c];
            Xa[i] = (_Float16)v;
        }
    }
    __syncthreads();
    layer_lds<DIN, 64>(Xa, Xb, Wl, Bl, wave, lane);
    __syncthreads();
    layer_lds<64, 64>(Xb, Xa, Wl + W1H, Bl + 64, wave, lane);
    __syncthreads();
#pragma unroll
    for (int nt = 0; nt < 2; ++nt) {
        v8f acc = wmma_rowtile<64, 32>(Xa, Wl + W1H + 4096, wave, lane, nt);
        float bv = Bl[128 + nt * 16 + (lane & 15)];
        int col = nt * 16 + (lane & 15);
        int mbase = wave * 16 + ((lane & 16) ? 8 : 0);
#pragma unroll
        for (int r = 0; r < 8; ++r) {
            int row = row0 + mbase + r;
            float y = sspf(acc[r] + bv);
            size_t o = (size_t)row * 32 + col;
            if (MODE == M_EDGE) {
                a.out[o] = a.resid[o] + y;
                int i0 = a.eidx[2 * row];
                atomicAdd(a.acc0 + (size_t)i0 * 32 + col, y);
                atomicAdd(a.acc1 + (size_t)(row / a.div) * 32 + col, y);
            } else if (MODE == M_NODE) {
                a.out[o] += y;
                atomicAdd(a.acc0 + (size_t)(row / a.div) * 32 + col, y);
            } else {
                a.out[o] += y;
            }
        }
    }
}

// ---------------------------------------------------------------------------
// Weight repack: f32 W[dout,din] -> f16 WMMA B-fragments, K padded to dinp.
// Fragment (kb, nt): 32 lanes x 16 halves; lane j: col = nt*16 + (j&15),
// k = kb*32 + (j>=16 ? 16 : 0) + h.
// ---------------------------------------------------------------------------
__global__ void pack_kernel(const float* __restrict__ W, _Float16* __restrict__ dst,
                            int din, int dinp, int dout)
{
    int idx = blockIdx.x * blockDim.x + threadIdx.x;
    int total = dinp * dout;
    if (idx >= total) return;
    int frag = idx >> 9;
    int within = idx & 511;
    int lane = within >> 4;
    int h = within & 15;
    int ntiles = dout >> 4;
    int kb = frag / ntiles;
    int nt = frag - kb * ntiles;
    int n = nt * 16 + (lane & 15);
    int k = kb * 32 + ((lane & 16) ? 16 : 0) + h;
    float v = (k < din) ? W[(size_t)n * din + k] : 0.0f;
    dst[idx] = (_Float16)v;
}

// ---------------------------------------------------------------------------
// Init kernels
// ---------------------------------------------------------------------------
__global__ void embed_kernel(const float* __restrict__ ae,
                             const float* __restrict__ w, const float* __restrict__ b,
                             float* __restrict__ out, int n_total)
{
    int t = blockIdx.x * blockDim.x + threadIdx.x;
    if (t >= n_total) return;
    int n = t >> 5, j = t & 31;
    float v = 0.0f;
    if (j < 16) {
        v = b[j];
        for (int k = 0; k < 13; ++k) v += w[j * 13 + k] * ae[(size_t)n * 13 + k];
    }
    out[t] = v;
}

__global__ void expand_kernel(const float* __restrict__ dist, float* __restrict__ out,
                              int E, float cutoff)
{
    int t = blockIdx.x * blockDim.x + threadIdx.x;
    if (t >= E * 32) return;
    int e = t >> 5, c = t & 31;
    float v = 0.0f;
    if (c < 16) {
        float d = dist[e];
        if (d < cutoff) {
            float env = 0.5f * (cosf(PI_F * d / cutoff) + 1.0f);
            float kk = (float)(c + 1);
            v = sinf(d * kk * PI_F / cutoff) / d * env;
        }
    }
    out[t] = v;
}

__global__ void u_init_kernel(const float* __restrict__ state, float* __restrict__ out,
                              int n_total)
{
    int t = blockIdx.x * blockDim.x + threadIdx.x;
    if (t >= n_total) return;
    out[t] = ((t & 31) == 0) ? state[t >> 5] : 0.0f;
}

// set2set with h=c=q*=0 and 1 iteration: q is a bias-only constant vector.
__global__ void s2s_q_kernel(const float* bih0, const float* bhh0,
                             const float* bih1, const float* bhh1,
                             const float* bih2, const float* bhh2,
                             float* __restrict__ qbuf)
{
    int tid = threadIdx.x;
    if (tid >= 96) return;
    int m = tid >> 5, t = tid & 31;
    const float* bih = (m == 0) ? bih0 : (m == 1) ? bih1 : bih2;
    const float* bhh = (m == 0) ? bhh0 : (m == 1) ? bhh1 : bhh2;
    float i_ = bih[t]      + bhh[t];
    float g_ = bih[64 + t] + bhh[64 + t];
    float o_ = bih[96 + t] + bhh[96 + t];
    float c  = sigmf(i_) * tanhf(g_);               // f-gate * 0 drops out
    qbuf[m * 32 + t] = sigmf(o_) * tanhf(c);
}

// Per-graph attention readout: out[g] = [q(32) | r(32)].
__global__ __launch_bounds__(256)
void s2s_attend(const float* __restrict__ x, const float* __restrict__ q,
                float* __restrict__ out, int S)
{
    int g = blockIdx.x, tid = threadIdx.x;
    __shared__ float e_lds[960];
    __shared__ float red[256];
    __shared__ float qs[32];
    const float* xg = x + (size_t)g * S * 32;
    if (tid < 32) qs[tid] = q[tid];
    __syncthreads();
    float lmax = -1e30f;
    for (int i = tid; i < S; i += 256) {
        float d = 0.0f;
        for (int c = 0; c < 32; ++c) d += xg[(size_t)i * 32 + c] * qs[c];
        e_lds[i] = d;
        lmax = fmaxf(lmax, d);
    }
    red[tid] = lmax; __syncthreads();
    for (int s = 128; s > 0; s >>= 1) {
        if (tid < s) red[tid] = fmaxf(red[tid], red[tid + s]);
        __syncthreads();
    }
    float m = red[0];
    __syncthreads();
    float lsum = 0.0f;
    for (int i = tid; i < S; i += 256) {
        float ex = expf(e_lds[i] - m);
        e_lds[i] = ex;
        lsum += ex;
    }
    red[tid] = lsum; __syncthreads();
    for (int s = 128; s > 0; s >>= 1) {
        if (tid < s) red[tid] += red[tid + s];
        __syncthreads();
    }
    float sum = red[0];
    __syncthreads();
    int c = tid & 31, grp = tid >> 5;
    float acc = 0.0f;
    for (int i = grp; i < S; i += 8) acc += e_lds[i] * xg[(size_t)i * 32 + c];
    red[tid] = acc; __syncthreads();
    if (tid < 32) {
        float r = 0.0f;
        for (int k = 0; k < 8; ++k) r += red[k * 32 + tid];
        out[(size_t)g * 64 + tid]      = qs[tid];
        out[(size_t)g * 64 + 32 + tid] = r / sum;
    }
}

__global__ void readout_kernel(const float* __restrict__ se, const float* __restrict__ sw,
                               const float* __restrict__ sn, const float* __restrict__ u,
                               const float* __restrict__ w0, const float* __restrict__ b0,
                               const float* __restrict__ w1, const float* __restrict__ b1,
                               const float* __restrict__ w2, const float* __restrict__ b2,
                               float* __restrict__ out, int nB)
{
    int g = blockIdx.x * blockDim.x + threadIdx.x;
    if (g >= nB) return;
    float in[224];
    for (int i = 0; i < 64; ++i) in[i]       = se[(size_t)g * 64 + i];
    for (int i = 0; i < 64; ++i) in[64 + i]  = sw[(size_t)g * 64 + i];
    for (int i = 0; i < 64; ++i) in[128 + i] = sn[(size_t)g * 64 + i];
    for (int i = 0; i < 32; ++i) in[192 + i] = u[(size_t)g * 32 + i];
    float h1[32];
    for (int o = 0; o < 32; ++o) {
        float a = b0[o];
        for (int k = 0; k < 224; ++k) a += w0[o * 224 + k] * in[k];
        h1[o] = sspf(a);
    }
    float h2[16];
    for (int o = 0; o < 16; ++o) {
        float a = b1[o];
        for (int k = 0; k < 32; ++k) a += w1[o * 32 + k] * h1[k];
        h2[o] = sspf(a);
    }
    float a = b2[0];
    for (int k = 0; k < 16; ++k) a += w2[k] * h2[k];
    out[g] = a;
}

// ---------------------------------------------------------------------------
// Host driver.
// Input leaf order (insertion order, depth-first):
//  0,1: atom_embedding {w,b}
//  2 + 52*blk: block weights:
//    +0 pre_ec(w0,b0,w1,b1) +4 pre_ev +8 pre_v +12 pre_u
//    +16 phi_ec(w0,b0,w1,b1,w2,b2) +22 phi_ev +28 phi_vc +34 phi_vv +40 phi_uc +46 phi_uv
//  158 se_che{w_ih,w_hh,b_ih,b_hh} 162 se_vdw 166 sv
//  170 readout (3 x {w,b})
//  176 atoms_embed 177 state 178 che_dist 179 vdw_dist 180 che_index
//  181 vdw_index 182 num_atoms 183 che_num_pairs 184 vdw_num_pairs
// ---------------------------------------------------------------------------
extern "C" void kernel_launch(void* const* d_in, const int* in_sizes, int n_in,
                              void* d_out, int out_size, void* d_ws, size_t ws_size,
                              hipStream_t stream)
{
    (void)in_sizes; (void)n_in; (void)out_size; (void)ws_size;
    constexpr int Bn = 1024, An = 40, Nn = Bn * An;
    constexpr int EC = Bn * 480, EV = Bn * 960;

    auto F = [&](int i) { return (const float*)d_in[i]; };
    auto I = [&](int i) { return (const int*)d_in[i]; };

    size_t off = 0;
    auto alloc = [&](size_t bytes) -> void* {
        void* p = (char*)d_ws + off;
        off += (bytes + 255) & ~(size_t)255;
        return p;
    };

    float* m_eche  = (float*)alloc((size_t)EC * 32 * 4);
    float* m_evdw  = (float*)alloc((size_t)EV * 32 * 4);
    float* m_nodes = (float*)alloc((size_t)Nn * 32 * 4);
    float* m_u     = (float*)alloc((size_t)Bn * 32 * 4);
    float* p_eche  = (float*)alloc((size_t)EC * 32 * 4);
    float* p_evdw  = (float*)alloc((size_t)EV * 32 * 4);
    float* p_v     = (float*)alloc((size_t)Nn * 32 * 4);
    float* p_u     = (float*)alloc((size_t)Bn * 32 * 4);
    float* ev_acc  = (float*)alloc((size_t)Nn * 32 * 4);
    float* eu_acc  = (float*)alloc((size_t)Bn * 32 * 4);
    float* vu_acc  = (float*)alloc((size_t)Bn * 32 * 4);
    float* qbuf    = (float*)alloc(3 * 32 * 4);
    float* se      = (float*)alloc((size_t)Bn * 64 * 4);
    float* sw      = (float*)alloc((size_t)Bn * 64 * 4);
    float* sn      = (float*)alloc((size_t)Bn * 64 * 4);

    auto pack = [&](const float* w, int din, int dinp, int dout) -> const _Float16* {
        _Float16* dst = (_Float16*)alloc((size_t)dinp * dout * 2);
        int total = dinp * dout;
        pack_kernel<<<(total + 255) / 256, 256, 0, stream>>>(w, dst, din, dinp, dout);
        return dst;
    };

    // --- pack all weights (stream-ordered, before first GEMM use) ---
    const _Float16* pw[3][26];
    const float*    bw[3][26];
    for (int blk = 0; blk < 3; ++blk) {
        int base = 2 + 52 * blk;
        int ein = blk ? 32 : 16, nin = blk ? 32 : 16, sin = blk ? 32 : 1;
        int pre_off[4] = {0, 4, 8, 12};
        int pre_din[4] = {ein, ein, nin, sin};
        for (int m = 0; m < 4; ++m) {
            pw[blk][m * 2 + 0] = pack(F(base + pre_off[m] + 0), pre_din[m], 32, 64);
            bw[blk][m * 2 + 0] = F(base + pre_off[m] + 1);
            pw[blk][m * 2 + 1] = pack(F(base + pre_off[m] + 2), 64, 64, 32);
            bw[blk][m * 2 + 1] = F(base + pre_off[m] + 3);
        }
        int phi_off[6] = {16, 22, 28, 34, 40, 46};
        int phi_din[6] = {128, 128, 96, 96, 96, 96};
        for (int m = 0; m < 6; ++m) {
            int o = base + phi_off[m];
            pw[blk][8 + m * 3 + 0] = pack(F(o + 0), phi_din[m], phi_din[m], 64);
            bw[blk][8 + m * 3 + 0] = F(o + 1);
            pw[blk][8 + m * 3 + 1] = pack(F(o + 2), 64, 64, 64);
            bw[blk][8 + m * 3 + 1] = F(o + 3);
            pw[blk][8 + m * 3 + 2] = pack(F(o + 4), 64, 64, 32);
            bw[blk][8 + m * 3 + 2] = F(o + 5);
        }
    }

    // --- init state ---
    embed_kernel<<<(Nn * 32 + 255) / 256, 256, 0, stream>>>(F(176), F(0), F(1), m_nodes, Nn * 32);
    expand_kernel<<<(EC * 32 + 255) / 256, 256, 0, stream>>>(F(178), m_eche, EC, 5.0f);
    expand_kernel<<<(EV * 32 + 255) / 256, 256, 0, stream>>>(F(179), m_evdw, EV, 5.0f);
    u_init_kernel<<<(Bn * 32 + 255) / 256, 256, 0, stream>>>(F(177), m_u, Bn * 32);

    const int* che_idx = I(180);
    const int* vdw_idx = I(181);

    // --- 3 message-passing blocks ---
    for (int blk = 0; blk < 3; ++blk) {
        mlp2_pre<32, 64, 32><<<EC / 128, 256, 0, stream>>>(
            m_eche, pw[blk][0], bw[blk][0], pw[blk][1], bw[blk][1], p_eche);
        mlp2_pre<32, 64, 32><<<EV / 128, 256, 0, stream>>>(
            m_evdw, pw[blk][2], bw[blk][2], pw[blk][3], bw[blk][3], p_evdw);
        mlp2_pre<32, 64, 32><<<Nn / 128, 256, 0, stream>>>(
            m_nodes, pw[blk][4], bw[blk][4], pw[blk][5], bw[blk][5], p_v);
        mlp2_pre<32, 64, 32><<<Bn / 128, 256, 0, stream>>>(
            m_u, pw[blk][6], bw[blk][6], pw[blk][7], bw[blk][7], p_u);

        hipMemcpyAsync(m_nodes, p_v, (size_t)Nn * 32 * 4, hipMemcpyDeviceToDevice, stream);
        hipMemcpyAsync(m_u, p_u, (size_t)Bn * 32 * 4, hipMemcpyDeviceToDevice, stream);

        for (int half = 0; half < 2; ++half) {
            hipMemsetAsync(ev_acc, 0, (size_t)Nn * 32 * 4, stream);
            hipMemsetAsync(eu_acc, 0, (size_t)Bn * 32 * 4, stream);
            hipMemsetAsync(vu_acc, 0, (size_t)Bn * 32 * 4, stream);

            int wb = half ? 11 : 8;   // phi_ev : phi_ec
            Mlp3Args ea{};
            ea.src0 = p_v; ea.src1 = half ? p_evdw : p_eche; ea.src2 = p_u;
            ea.eidx = half ? vdw_idx : che_idx; ea.div = half ? 960 : 480;
            ea.w1 = pw[blk][wb];     ea.b1 = bw[blk][wb];
            ea.w2 = pw[blk][wb + 1]; ea.b2 = bw[blk][wb + 1];
            ea.w3 = pw[blk][wb + 2]; ea.b3 = bw[blk][wb + 2];
            ea.out = half ? m_evdw : m_eche;
            ea.resid = half ? p_evdw : p_eche;
            ea.acc0 = ev_acc; ea.acc1 = eu_acc;
            mlp3_kernel<M_EDGE, 128><<<(half ? EV : EC) / 128, 256, 0, stream>>>(ea);

            int nb = half ? 17 : 14;  // phi_vv : phi_vc
            Mlp3Args na{};
            na.src0 = ev_acc; na.src1 = p_v; na.src2 = p_u; na.div = An;
            na.w1 = pw[blk][nb];     na.b1 = bw[blk][nb];
            na.w2 = pw[blk][nb + 1]; na.b2 = bw[blk][nb + 1];
            na.w3 = pw[blk][nb + 2]; na.b3 = bw[blk][nb + 2];
            na.out = m_nodes; na.acc0 = vu_acc;
            mlp3_kernel<M_NODE, 96><<<Nn / 128, 256, 0, stream>>>(na);

            int ub = half ? 23 : 20;  // phi_uv : phi_uc
            Mlp3Args ua{};
            ua.src0 = eu_acc; ua.src1 = vu_acc; ua.src2 = p_u; ua.div = 1;
            ua.w1 = pw[blk][ub];     ua.b1 = bw[blk][ub];
            ua.w2 = pw[blk][ub + 1]; ua.b2 = bw[blk][ub + 1];
            ua.w3 = pw[blk][ub + 2]; ua.b3 = bw[blk][ub + 2];
            ua.out = m_u;
            mlp3_kernel<M_U, 96><<<Bn / 128, 256, 0, stream>>>(ua);
        }
    }

    // --- set2set readouts + MLP head ---
    s2s_q_kernel<<<1, 96, 0, stream>>>(F(160), F(161), F(164), F(165), F(168), F(169), qbuf);
    s2s_attend<<<Bn, 256, 0, stream>>>(m_eche, qbuf + 0,  se, 480);
    s2s_attend<<<Bn, 256, 0, stream>>>(m_evdw, qbuf + 32, sw, 960);
    s2s_attend<<<Bn, 256, 0, stream>>>(m_nodes, qbuf + 64, sn, An);
    readout_kernel<<<(Bn + 255) / 256, 256, 0, stream>>>(
        se, sw, sn, m_u, F(170), F(171), F(172), F(173), F(174), F(175),
        (float*)d_out, Bn);
}
